// Block_78615081386225
// MI455X (gfx1250) — matrix operand order repaired and checked by reading, not compiled, not run on hardware
//
#include <hip/hip_runtime.h>
#include <hip/hip_bf16.h>
#include <math.h>

typedef __attribute__((ext_vector_type(16))) _Float16 v16h;
typedef __attribute__((ext_vector_type(8)))  _Float16 h8;
typedef __attribute__((ext_vector_type(4)))  _Float16 h4;
typedef __attribute__((ext_vector_type(8)))  float    v8f;

#define WMMA_F16(a, b, c) \
  __builtin_amdgcn_wmma_f32_16x16x32_f16(false, (a), false, (b), (short)0, (c), false, false)
#define CAT16(a, b) \
  __builtin_shufflevector((a), (b), 0,1,2,3,4,5,6,7,8,9,10,11,12,13,14,15)

// ---------------------------------------------------------------------------
// f32 -> f16 bulk convert (4 elems/thread, vectorized). total4 = elems/4.
// ---------------------------------------------------------------------------
__global__ void __launch_bounds__(256)
cvt16_k(const float* __restrict__ s, _Float16* __restrict__ d, int total4)
{
  const int i = blockIdx.x * 256 + threadIdx.x;
  if (i >= total4) return;
  const float4 v = ((const float4*)s)[i];
  h4 o;
  o[0] = (_Float16)v.x; o[1] = (_Float16)v.y;
  o[2] = (_Float16)v.z; o[3] = (_Float16)v.w;
  ((h4*)d)[i] = o;
}

// ---------------------------------------------------------------------------
// Tiled WMMA GEMM: C[b] = alpha * (A[b] @ B[b]) + bias + Res[b]
// A: M x K f16 (lda), B: K x N f16 (ldb), C: M x N f32 (ldc).
// One wave owns a 32(M) x 16(N) tile (2 accumulators share one B fragment);
// block = 8 waves -> 32 x 128 super-tile. M%32==0, N%128==0, K%32==0,
// lda/ldb multiples of 8 (16B-aligned b128 fragment loads, zero cvt in loop).
// ---------------------------------------------------------------------------
__global__ void __launch_bounds__(256)
gemm_wmma_f16(const _Float16* __restrict__ A, const _Float16* __restrict__ B,
              float* __restrict__ C, const float* __restrict__ bias,
              const float* __restrict__ Res,
              int M, int K, int N, int lda, int ldb, int ldc, int ldres,
              long long sA, long long sB, long long sC, long long sRes,
              float alpha)
{
  const int wave = threadIdx.x >> 5;
  const int lane = threadIdx.x & 31;
  const int n0 = blockIdx.x * 128 + wave * 16;
  const int m0 = blockIdx.y * 32;
  const int bz = blockIdx.z;

  const _Float16* Ab = A + (size_t)bz * sA;
  const _Float16* Bb = B + (size_t)bz * sB;
  float*          Cb = C + (size_t)bz * sC;
  const float*    Rb = Res ? (Res + (size_t)bz * sRes) : nullptr;

  const int m = lane & 15;
  const int h = lane >> 4;

  const v8f vz = {0.f,0.f,0.f,0.f,0.f,0.f,0.f,0.f};
  v8f acc0 = vz, acc1 = vz;
  // A fragment rows: lane row m (and m+16 for second tile); halves h*8 select K groups.
  const _Float16* a0 = Ab + (size_t)(m0 + m) * lda + h * 8;
  const _Float16* a1 = a0 + (size_t)16 * lda;

  for (int k0 = 0; k0 < K; k0 += 32) {
    const h8 p0 = *(const h8*)(a0 + k0);
    const h8 p1 = *(const h8*)(a0 + k0 + 16);
    const h8 p2 = *(const h8*)(a1 + k0);
    const h8 p3 = *(const h8*)(a1 + k0 + 16);
    const _Float16* bp = Bb + (size_t)(k0 + h * 8) * ldb + n0 + m;  // B col n0+m, K table order
    v16h bf;
#pragma unroll
    for (int e = 0; e < 16; ++e) bf[e] = bp[(size_t)((e & 7) + ((e >> 3) << 4)) * ldb];
    if (k0 + 32 < K) __builtin_prefetch(bp + (size_t)32 * ldb, 0, 1); // global_prefetch_b8
    const v16h af0 = CAT16(p0, p1);
    const v16h af1 = CAT16(p2, p3);

    acc0 = WMMA_F16(af0, bf, acc0);
    acc1 = WMMA_F16(af1, bf, acc1);
  }

  const int col = n0 + m;
  const float bv = bias ? bias[col] : 0.f;
#pragma unroll
  for (int r = 0; r < 8; ++r) {
    int row = m0 + r + 8 * h;                   // C/D layout: VGPR r -> row r+8h
    float v = acc0[r] * alpha + bv;
    if (Rb) v += Rb[(size_t)row * ldres + col];
    Cb[(size_t)row * ldc + col] = v;
    row += 16;
    v = acc1[r] * alpha + bv;
    if (Rb) v += Rb[(size_t)row * ldres + col];
    Cb[(size_t)row * ldc + col] = v;
  }
}

// ---------------------------------------------------------------------------
// RMSNorm over last dim; f32 in, f16 out (feeds WMMA GEMMs directly).
// ---------------------------------------------------------------------------
__global__ void __launch_bounds__(256)
rmsnorm_k(const float* __restrict__ x, const float* __restrict__ w,
          _Float16* __restrict__ out, int D)
{
  __shared__ float red[256];
  const int row = blockIdx.x, tid = threadIdx.x;
  const float* xr = x + (size_t)row * D;
  float s = 0.f;
  for (int d = tid; d < D; d += 256) { float v = xr[d]; s += v * v; }
  red[tid] = s; __syncthreads();
  for (int st = 128; st > 0; st >>= 1) { if (tid < st) red[tid] += red[tid + st]; __syncthreads(); }
  const float scale = rsqrtf(red[0] / (float)D + 1e-6f);
  _Float16* orow = out + (size_t)row * D;
  for (int d = tid; d < D; d += 256) orow[d] = (_Float16)(xr[d] * scale * w[d]);
}

// ---------------------------------------------------------------------------
// RoPE on q/k + v copy; f32 qkv in, f16 out (B,H,T,64)/(B,4,T,64).
// ---------------------------------------------------------------------------
__global__ void __launch_bounds__(256)
rope_k(const float* __restrict__ qkv, const float* __restrict__ fcos,
       const float* __restrict__ fsin, _Float16* __restrict__ qr,
       _Float16* __restrict__ kr, _Float16* __restrict__ vr, int total)
{
  int idx = blockIdx.x * 256 + threadIdx.x;
  if (idx >= total) return;
  const int i  = idx & 31;
  const int ha = (idx >> 5) % 24;
  const int n  = idx / (32 * 24);
  const int t  = n & 1023;
  const int b  = n >> 10;
  const float c = fcos[t * 32 + i], s = fsin[t * 32 + i];
  if (ha < 16) {
    const float* p = qkv + (size_t)n * 1536 + ha * 64 + 2 * i;
    const float a = p[0], bb = p[1];
    _Float16* o = qr + ((((size_t)b * 16 + ha) * 1024 + t) * 64) + 2 * i;
    o[0] = (_Float16)(a * c - bb * s); o[1] = (_Float16)(a * s + bb * c);
  } else if (ha < 20) {
    const int hk = ha - 16;
    const float* p = qkv + (size_t)n * 1536 + 1024 + hk * 64 + 2 * i;
    const float a = p[0], bb = p[1];
    _Float16* o = kr + ((((size_t)b * 4 + hk) * 1024 + t) * 64) + 2 * i;
    o[0] = (_Float16)(a * c - bb * s); o[1] = (_Float16)(a * s + bb * c);
  } else {
    const int hv = ha - 20;
    const float* p = qkv + (size_t)n * 1536 + 1280 + hv * 64 + 2 * i;
    _Float16* o = vr + ((((size_t)b * 4 + hv) * 1024 + t) * 64) + 2 * i;
    o[0] = (_Float16)p[0]; o[1] = (_Float16)p[1];
  }
}

// ---------------------------------------------------------------------------
// Flash attention, WMMA. One wave per (b, head, 16-query tile), 32 keys/iter.
// f16 Q/K/V; K/V staged to LDS; P routed through LDS into full 16x32 A frag.
// 8 WMMAs per 32 keys (4 score + 4 PV), no zero-padding waste.
// ---------------------------------------------------------------------------
__global__ void __launch_bounds__(32)
attn_flash(const _Float16* __restrict__ q, const _Float16* __restrict__ k,
           const _Float16* __restrict__ v, _Float16* __restrict__ y)
{
  __shared__ _Float16 kt[32 * 64];
  __shared__ _Float16 vt[32 * 64];
  __shared__ _Float16 pt[16 * 32];
  const int lane = threadIdx.x;
  const int q0 = blockIdx.x * 16;
  const int hh = blockIdx.y;
  const int b  = blockIdx.z;
  const int kvh = hh >> 2;                         // GQA rep = 4
  const _Float16* qp = q + (((size_t)b * 16 + hh) * 1024) * 64;
  const _Float16* kp = k + (((size_t)b * 4 + kvh) * 1024) * 64;
  const _Float16* vp = v + (((size_t)b * 4 + kvh) * 1024) * 64;
  const int m = lane & 15, h = lane >> 4;

  v16h qf0, qf1;
  {
    const _Float16* ar = qp + (size_t)(q0 + m) * 64 + h * 8;
    const h8 a0 = *(const h8*)(ar);
    const h8 a1 = *(const h8*)(ar + 16);
    const h8 a2 = *(const h8*)(ar + 32);
    const h8 a3 = *(const h8*)(ar + 48);
    qf0 = CAT16(a0, a1);
    qf1 = CAT16(a2, a3);
  }
  float rm[8], rl[8];
#pragma unroll
  for (int r = 0; r < 8; ++r) { rm[r] = -3.0e38f; rl[r] = 0.f; }
  const v8f vz = {0.f,0.f,0.f,0.f,0.f,0.f,0.f,0.f};
  v8f o[4] = {vz, vz, vz, vz};

  const unsigned* kp32 = (const unsigned*)kp;
  const unsigned* vp32 = (const unsigned*)vp;
  unsigned* kt32 = (unsigned*)kt;
  unsigned* vt32 = (unsigned*)vt;

  for (int j0 = 0; j0 <= q0; j0 += 32) {
    for (int i = lane; i < 32 * 32; i += 32) {     // 32 keys x 64 halves (dword copies)
      const int krw = i >> 5, c2 = i & 31;
      const int key = j0 + krw;
      unsigned kv = 0u, vv = 0u;
      if (key < 1024) { kv = kp32[(size_t)key * 32 + c2]; vv = vp32[(size_t)key * 32 + c2]; }
      kt32[i] = kv; vt32[i] = vv;
    }
    __syncthreads();
    v8f sa = vz, sb = vz;                          // keys j0..+15 / j0+16..+31
    {
      v16h bf;
#pragma unroll
      for (int e = 0; e < 16; ++e) bf[e] = kt[(size_t)m * 64 + h * 8 + (e & 7) + ((e >> 3) << 4)];
      sa = WMMA_F16(qf0, bf, sa);
#pragma unroll
      for (int e = 0; e < 16; ++e) bf[e] = kt[(size_t)m * 64 + 32 + h * 8 + (e & 7) + ((e >> 3) << 4)];
      sa = WMMA_F16(qf1, bf, sa);
#pragma unroll
      for (int e = 0; e < 16; ++e) bf[e] = kt[(size_t)(16 + m) * 64 + h * 8 + (e & 7) + ((e >> 3) << 4)];
      sb = WMMA_F16(qf0, bf, sb);
#pragma unroll
      for (int e = 0; e < 16; ++e) bf[e] = kt[(size_t)(16 + m) * 64 + 32 + h * 8 + (e & 7) + ((e >> 3) << 4)];
      sb = WMMA_F16(qf1, bf, sb);
    }
#pragma unroll
    for (int r = 0; r < 8; ++r) {
      const int qi = q0 + r + 8 * h;
      float va = sa[r] * 0.125f;                   // 1/sqrt(64)
      float vb = sb[r] * 0.125f;
      if (j0 + m > qi)      va = -3.0e38f;         // causal mask
      if (j0 + 16 + m > qi) vb = -3.0e38f;
      float mx = fmaxf(va, vb);
#pragma unroll
      for (int mk = 8; mk; mk >>= 1) mx = fmaxf(mx, __shfl_xor(mx, mk, 32));
      const float nm = fmaxf(rm[r], mx);
      const float pa = __expf(va - nm);
      const float pb = __expf(vb - nm);
      const float al = __expf(rm[r] - nm);
      float ps = pa + pb;
#pragma unroll
      for (int mk = 8; mk; mk >>= 1) ps += __shfl_xor(ps, mk, 32);
      rl[r] = rl[r] * al + ps;
      rm[r] = nm;
#pragma unroll
      for (int dt = 0; dt < 4; ++dt) o[dt][r] *= al;
      pt[(r + 8 * h) * 32 + m]      = (_Float16)pa;
      pt[(r + 8 * h) * 32 + 16 + m] = (_Float16)pb;
    }
    __syncthreads();
    v16h pf;                                       // full 16x32 A fragment of P
#pragma unroll
    for (int e = 0; e < 8; ++e) {
      pf[e]     = pt[m * 32 + h * 8 + e];          // K = h*8 + e
      pf[8 + e] = pt[m * 32 + h * 8 + 16 + e];     // K = h*8 + 16 + e
    }
#pragma unroll
    for (int dt = 0; dt < 4; ++dt) {
      v16h vf;
#pragma unroll
      for (int e = 0; e < 16; ++e) vf[e] = vt[(size_t)(h * 8 + (e & 7) + ((e >> 3) << 4)) * 64 + dt * 16 + m];
      o[dt] = WMMA_F16(pf, vf, o[dt]);
    }
    __syncthreads();
  }
#pragma unroll
  for (int r = 0; r < 8; ++r) {
    const float inv = 1.0f / rl[r];
    const int row = q0 + r + 8 * h;
#pragma unroll
    for (int dt = 0; dt < 4; ++dt)
      y[((size_t)b * 1024 + row) * 1024 + hh * 64 + dt * 16 + m] = (_Float16)(o[dt][r] * inv);
  }
}

// ---------------------------------------------------------------------------
// Router: logits = xn @ gate_w (N x 8). Block per token, wave e -> expert e.
// ---------------------------------------------------------------------------
__global__ void __launch_bounds__(256)
gate_k(const _Float16* __restrict__ xn, const float* __restrict__ gw, float* __restrict__ logits)
{
  const int n = blockIdx.x;
  const int e = threadIdx.x >> 5, lane = threadIdx.x & 31;
  const _Float16* xr = xn + (size_t)n * 1024;
  float s = 0.f;
  for (int d = lane; d < 1024; d += 32) s += (float)xr[d] * gw[d * 8 + e];
#pragma unroll
  for (int mk = 16; mk; mk >>= 1) s += __shfl_xor(s, mk, 32);
  if (lane == 0) logits[n * 8 + e] = s;
}

__global__ void __launch_bounds__(256)
topk_k(const float* __restrict__ logits, float* __restrict__ tvals,
       int* __restrict__ tidx, float* __restrict__ tprobs, int N)
{
  const int n = blockIdx.x * 256 + threadIdx.x;
  if (n >= N) return;
  float l[8];
#pragma unroll
  for (int e = 0; e < 8; ++e) l[e] = logits[n * 8 + e];
  int i1 = 0;
#pragma unroll
  for (int e = 1; e < 8; ++e) if (l[e] > l[i1]) i1 = e;
  int i2 = (i1 == 0) ? 1 : 0;
#pragma unroll
  for (int e = 0; e < 8; ++e) if (e != i1 && l[e] > l[i2]) i2 = e;
  const float v1 = l[i1], v2 = l[i2];
  const float p1 = 1.f / (1.f + __expf(v2 - v1));
  tvals[2*n] = v1; tvals[2*n+1] = v2;
  tidx[2*n]  = i1; tidx[2*n+1]  = i2;
  tprobs[2*n] = p1; tprobs[2*n+1] = 1.f - p1;
}

// Deterministic aux-loss reduction (single block, fixed-order tree).
__global__ void __launch_bounds__(256)
aux_k(const float* __restrict__ logits, const int* __restrict__ tidx,
      const float* __restrict__ tprobs, float* __restrict__ auxout, int N)
{
  __shared__ float red[17][256];
  const int tid = threadIdx.x;
  float racc[8] = {0,0,0,0,0,0,0,0}, aacc[8] = {0,0,0,0,0,0,0,0}, zacc = 0.f;
  for (int n = tid; n < N; n += 256) {
    const float* l = logits + n * 8;
    float mx = l[0];
#pragma unroll
    for (int e = 1; e < 8; ++e) mx = fmaxf(mx, l[e]);
    float ex[8], sum = 0.f;
#pragma unroll
    for (int e = 0; e < 8; ++e) { ex[e] = __expf(l[e] - mx); sum += ex[e]; }
    const float inv = 1.f / sum;
#pragma unroll
    for (int e = 0; e < 8; ++e) racc[e] += ex[e] * inv;
    const float zl = mx + logf(sum);
    zacc += zl * zl;
    aacc[tidx[2*n]]   += tprobs[2*n];
    aacc[tidx[2*n+1]] += tprobs[2*n+1];
  }
#pragma unroll
  for (int e = 0; e < 8; ++e) { red[e][tid] = racc[e]; red[8+e][tid] = aacc[e]; }
  red[16][tid] = zacc;
  __syncthreads();
  for (int st = 128; st > 0; st >>= 1) {
    if (tid < st)
      for (int qn = 0; qn < 17; ++qn) red[qn][tid] += red[qn][tid + st];
    __syncthreads();
  }
  if (tid == 0) {
    float bal = 0.f;
    const float invN = 1.f / (float)N;
    for (int e = 0; e < 8; ++e) bal += (red[e][0] * invN) * (red[8+e][0] * invN);
    bal *= 8.f;
    auxout[0] = 0.01f * bal + 0.001f * (red[16][0] * invN);
  }
}

// Exact lexsort rank (expert asc, prio desc, stable): O(NK^2) cmps from LDS.
__global__ void __launch_bounds__(256)
slot_k(const int* __restrict__ tidx, const float* __restrict__ tvals,
       int* __restrict__ slot, int* __restrict__ keep, int NK, int Cap)
{
  __shared__ int   se[4096];
  __shared__ float sp[4096];
  for (int i = threadIdx.x; i < NK; i += 256) { se[i] = tidx[i]; sp[i] = tvals[i]; }
  __syncthreads();
  const int i = blockIdx.x * 256 + threadIdx.x;
  if (i >= NK) return;
  const int e = se[i]; const float p = sp[i];
  int rank = 0;
  for (int j = 0; j < NK; ++j)
    if (se[j] == e && (sp[j] > p || (sp[j] == p && j < i))) ++rank;
  slot[i] = rank;
  keep[i] = (rank < Cap) ? 1 : 0;
}

// Scatter token rows into capacity buffers: f32 (residual) + f16 (GEMM A).
__global__ void __launch_bounds__(256)
scatter_k(const _Float16* __restrict__ xn, const int* __restrict__ tidx,
          const int* __restrict__ slot, const int* __restrict__ keep,
          float* __restrict__ buf, _Float16* __restrict__ bufh, int Cap)
{
  const int i = blockIdx.x;
  if (!keep[i]) return;
  const int d = blockIdx.y * 256 + threadIdx.x;
  const int n = i >> 1;
  const _Float16 v = xn[(size_t)n * 1024 + d];
  const size_t o = ((size_t)tidx[i] * Cap + slot[i]) * 1024 + d;
  buf[o]  = (float)v;
  bufh[o] = v;
}

__global__ void __launch_bounds__(256)
silu_mul_k(const float* __restrict__ gu, _Float16* __restrict__ act, int total)
{
  const int idx = blockIdx.x * 256 + threadIdx.x;
  if (idx >= total) return;
  const int row = idx / 2816, j = idx % 2816;
  const size_t base = (size_t)row * 5632;
  const float g = gu[base + j], u = gu[base + 2816 + j];
  act[(size_t)row * 2816 + j] = (_Float16)((g / (1.f + __expf(-g))) * u);
}

__global__ void __launch_bounds__(256)
build_seq_k(const float* __restrict__ med, const float* __restrict__ eout,
            const int* __restrict__ tidx, const int* __restrict__ slot,
            const int* __restrict__ keep, float* __restrict__ seq, int Cap, int N)
{
  const int idx = blockIdx.x * 256 + threadIdx.x;
  if (idx >= N * 1024) return;
  const int n = idx >> 10, d = idx & 1023;
  seq[((size_t)n * 3) * 1024 + d] = med[d];
#pragma unroll
  for (int kk = 0; kk < 2; ++kk) {
    const int i = 2 * n + kk;
    float v = 0.f;
    if (keep[i]) {
      int s = slot[i]; if (s > Cap - 1) s = Cap - 1;
      v = eout[((size_t)tidx[i] * Cap + s) * 1024 + d];
    }
    seq[((size_t)n * 3 + 1 + kk) * 1024 + d] = v;
  }
}

// 3-token collaboration attention, 4 heads of 256 dims, no mask; f16 out.
__global__ void __launch_bounds__(256)
collab_attn_k(const float* __restrict__ sq, const float* __restrict__ sk,
              const float* __restrict__ sv, _Float16* __restrict__ so, int N)
{
  const int idx = blockIdx.x * 256 + threadIdx.x;
  if (idx >= N * 12) return;
  const int qi = idx % 3;
  const int hh = (idx / 3) & 3;
  const int n  = idx / 12;
  const float* qr = sq + ((size_t)(n * 3 + qi)) * 1024 + hh * 256;
  float s[3];
#pragma unroll
  for (int j = 0; j < 3; ++j) {
    const float* krp = sk + ((size_t)(n * 3 + j)) * 1024 + hh * 256;
    float acc = 0.f;
    for (int d = 0; d < 256; ++d) acc += qr[d] * krp[d];
    s[j] = acc * 0.0625f;                          // 1/sqrt(256)
  }
  const float mx = fmaxf(s[0], fmaxf(s[1], s[2]));
  float e0 = __expf(s[0] - mx), e1 = __expf(s[1] - mx), e2 = __expf(s[2] - mx);
  const float inv = 1.f / (e0 + e1 + e2);
  e0 *= inv; e1 *= inv; e2 *= inv;
  const float* v0 = sv + ((size_t)(n * 3)) * 1024 + hh * 256;
  const float* v1 = v0 + 1024;
  const float* v2 = v1 + 1024;
  _Float16* op = so + ((size_t)(n * 3 + qi)) * 1024 + hh * 256;
  for (int d = 0; d < 256; ++d)
    op[d] = (_Float16)(e0 * v0[d] + e1 * v1[d] + e2 * v2[d]);
}

__global__ void __launch_bounds__(256)
gelu_k(const float* __restrict__ x, _Float16* __restrict__ out, int total)
{
  const int idx = blockIdx.x * 256 + threadIdx.x;
  if (idx >= total) return;
  const float v = x[idx];
  out[idx] = (_Float16)(0.5f * v * (1.f + erff(v * 0.70710678118654752f)));
}

__global__ void __launch_bounds__(256)
gsel_k(const float* __restrict__ seq, const float* __restrict__ fgw,
       const float* __restrict__ fgb, float* __restrict__ gsel)
{
  __shared__ float red[256];
  const int n = blockIdx.x, tid = threadIdx.x;
  const float* medp = seq + ((size_t)n * 3) * 1024;
  float s = 0.f;
  for (int d = tid; d < 1024; d += 256) s += medp[d] * fgw[d];
  red[tid] = s; __syncthreads();
  for (int st = 128; st > 0; st >>= 1) { if (tid < st) red[tid] += red[tid + st]; __syncthreads(); }
  if (tid == 0) gsel[n] = 1.f / (1.f + __expf(-(red[0] + fgb[0])));
}

__global__ void __launch_bounds__(256)
fuse_k(const float* __restrict__ seq, const float* __restrict__ tprobs,
       const float* __restrict__ gsel, _Float16* __restrict__ fused, int N)
{
  const int idx = blockIdx.x * 256 + threadIdx.x;
  if (idx >= N * 1024) return;
  const int n = idx >> 10, d = idx & 1023;
  const float medv = seq[((size_t)n * 3) * 1024 + d];
  const float w = tprobs[2*n]   * seq[((size_t)n * 3 + 1) * 1024 + d]
                + tprobs[2*n+1] * seq[((size_t)n * 3 + 2) * 1024 + d];
  const float g = gsel[n];
  fused[(size_t)n * 1024 + d] = (_Float16)(g * medv + (1.f - g) * w);
}

// ---------------------------------------------------------------------------
extern "C" void kernel_launch(void* const* d_in, const int* in_sizes, int n_in,
                              void* d_out, int out_size, void* d_ws, size_t ws_size,
                              hipStream_t stream)
{
  (void)in_sizes; (void)n_in; (void)out_size; (void)ws_size;
  constexpr int Bc = 2, Tc = 1024, Dd = 1024, Hh = 16, HID = 2816;
  constexpr int N = Bc * Tc;        // 2048
  constexpr int NK = 2 * N;         // 4096
  constexpr int Cap = 1024;         // capacity per expert
  constexpr int N3 = 3 * N;         // 6144 collab rows

  const float* x     = (const float*)d_in[0];
  const float* fcos  = (const float*)d_in[1];
  const float* fsin  = (const float*)d_in[2];
  const float* ln1   = (const float*)d_in[3];
  const float* ln2   = (const float*)d_in[4];
  const float* wqkv  = (const float*)d_in[5];
  const float* wo    = (const float*)d_in[6];
  const float* gatew = (const float*)d_in[7];
  const float* w13   = (const float*)d_in[8];
  const float* w2    = (const float*)d_in[9];
  const float* medi  = (const float*)d_in[10];
  const float* mwq   = (const float*)d_in[11];
  const float* mwk   = (const float*)d_in[12];
  const float* mwv   = (const float*)d_in[13];
  const float* mbq   = (const float*)d_in[14];
  const float* mbk   = (const float*)d_in[15];
  const float* mbv   = (const float*)d_in[16];
  const float* mwo   = (const float*)d_in[17];
  const float* mbo   = (const float*)d_in[18];
  const float* cn1   = (const float*)d_in[19];
  const float* cn2   = (const float*)d_in[20];
  const float* cf1   = (const float*)d_in[21];
  const float* cf2   = (const float*)d_in[22];
  const float* fgw   = (const float*)d_in[23];
  const float* fgb   = (const float*)d_in[24];
  const float* moeo  = (const float*)d_in[25];
  float* out = (float*)d_out;

  size_t off = 0;
  auto allocB = [&](size_t bytes) -> void* {
    void* p = (char*)d_ws + off;
    off += (bytes + 255) & ~(size_t)255;
    return p;
  };
  auto f32buf = [&](size_t elems) { return (float*)allocB(elems * 4); };
  auto f16buf = [&](size_t elems) { return (_Float16*)allocB(elems * 2); };
  auto i32buf = [&](size_t elems) { return (int*)allocB(elems * 4); };

  // f16 weight mirrors (converted once per launch; ~450MB traffic ≈ 20us @23.3TB/s)
  _Float16* wqkvh = f16buf((size_t)Dd * 1536);
  _Float16* woh   = f16buf((size_t)Dd * Dd);
  _Float16* w13h  = f16buf((size_t)8 * Dd * 2 * HID);
  _Float16* w2h   = f16buf((size_t)8 * HID * Dd);
  _Float16* mwqh  = f16buf((size_t)Dd * Dd);
  _Float16* mwkh  = f16buf((size_t)Dd * Dd);
  _Float16* mwvh  = f16buf((size_t)Dd * Dd);
  _Float16* mwoh  = f16buf((size_t)Dd * Dd);
  _Float16* cf1h  = f16buf((size_t)Dd * Dd);
  _Float16* cf2h  = f16buf((size_t)Dd * Dd);
  _Float16* moeoh = f16buf((size_t)Dd * Dd);
  // activations
  _Float16* normh = f16buf((size_t)N * Dd);
  float*    qkv   = f32buf((size_t)N * 1536);
  _Float16* qrh   = f16buf((size_t)N * Dd);
  _Float16* krh   = f16buf((size_t)N * 256);
  _Float16* vrh   = f16buf((size_t)N * 256);
  _Float16* yh    = f16buf((size_t)N * Dd);
  float*    x1    = f32buf((size_t)N * Dd);
  float*    logits= f32buf((size_t)N * 8);
  float*    tvals = f32buf((size_t)NK);
  float*    tprobs= f32buf((size_t)NK);
  int*      tidx  = i32buf((size_t)NK);
  int*      slotA = i32buf((size_t)NK);
  int*      keepA = i32buf((size_t)NK);
  float*    buf   = f32buf((size_t)8 * Cap * Dd);
  _Float16* bufh  = f16buf((size_t)8 * Cap * Dd);
  float*    gu    = f32buf((size_t)8 * Cap * 2 * HID);
  _Float16* acth  = f16buf((size_t)8 * Cap * HID);
  float*    eout  = f32buf((size_t)8 * Cap * Dd);
  float*    seq   = f32buf((size_t)N3 * Dd);
  _Float16* snormh= f16buf((size_t)N3 * Dd);
  float*    sqb   = f32buf((size_t)N3 * Dd);
  float*    skb   = f32buf((size_t)N3 * Dd);
  float*    svb   = f32buf((size_t)N3 * Dd);
  _Float16* sobh  = f16buf((size_t)N3 * Dd);
  _Float16* sqh   = f16buf((size_t)N3 * Dd);
  _Float16* fusedh= f16buf((size_t)N * Dd);
  float*    gsel  = f32buf((size_t)N);

  auto cvt = [&](const float* s, _Float16* d, size_t elems) {
    const int t4 = (int)(elems / 4);
    cvt16_k<<<(t4 + 255) / 256, 256, 0, stream>>>(s, d, t4);
  };
  cvt(wqkv, wqkvh, (size_t)Dd * 1536);
  cvt(wo,   woh,   (size_t)Dd * Dd);
  cvt(w13,  w13h,  (size_t)8 * Dd * 2 * HID);
  cvt(w2,   w2h,   (size_t)8 * HID * Dd);
  cvt(mwq,  mwqh,  (size_t)Dd * Dd);
  cvt(mwk,  mwkh,  (size_t)Dd * Dd);
  cvt(mwv,  mwvh,  (size_t)Dd * Dd);
  cvt(mwo,  mwoh,  (size_t)Dd * Dd);
  cvt(cf1,  cf1h,  (size_t)Dd * Dd);
  cvt(cf2,  cf2h,  (size_t)Dd * Dd);
  cvt(moeo, moeoh, (size_t)Dd * Dd);

  // --- attention block ---
  rmsnorm_k<<<N, 256, 0, stream>>>(x, ln1, normh, Dd);
  gemm_wmma_f16<<<dim3(1536/128, N/32, 1), 256, 0, stream>>>(
      normh, wqkvh, qkv, nullptr, nullptr, N, Dd, 1536, Dd, 1536, 1536, 0, 0,0,0,0, 1.f);
  rope_k<<<(N*24*32 + 255)/256, 256, 0, stream>>>(qkv, fcos, fsin, qrh, krh, vrh, N*24*32);
  attn_flash<<<dim3(Tc/16, Hh, Bc), 32, 0, stream>>>(qrh, krh, vrh, yh);
  gemm_wmma_f16<<<dim3(Dd/128, N/32, 1), 256, 0, stream>>>(
      yh, woh, x1, nullptr, x, N, Dd, Dd, Dd, Dd, Dd, Dd, 0,0,0,0, 1.f);

  // --- routing ---
  rmsnorm_k<<<N, 256, 0, stream>>>(x1, ln2, normh, Dd);
  gate_k<<<N, 256, 0, stream>>>(normh, gatew, logits);
  topk_k<<<(N + 255)/256, 256, 0, stream>>>(logits, tvals, tidx, tprobs, N);
  aux_k<<<1, 256, 0, stream>>>(logits, tidx, tprobs, out + (size_t)N * Dd, N);
  slot_k<<<NK/256, 256, 0, stream>>>(tidx, tvals, slotA, keepA, NK, Cap);
  hipMemsetAsync(buf,  0, (size_t)8 * Cap * Dd * sizeof(float), stream);
  hipMemsetAsync(bufh, 0, (size_t)8 * Cap * Dd * sizeof(_Float16), stream);
  scatter_k<<<dim3(NK, Dd/256), 256, 0, stream>>>(normh, tidx, slotA, keepA, buf, bufh, Cap);

  // --- expert FFN: gu = (2*buf) @ w13 ; eout = buf + (silu(g)*u) @ w2 ---
  gemm_wmma_f16<<<dim3((2*HID)/128, Cap/32, 8), 256, 0, stream>>>(
      bufh, w13h, gu, nullptr, nullptr, Cap, Dd, 2*HID, Dd, 2*HID, 2*HID, 0,
      (long long)Cap*Dd, (long long)Dd*2*HID, (long long)Cap*2*HID, 0, 2.f);
  silu_mul_k<<<((size_t)8*Cap*HID + 255)/256, 256, 0, stream>>>(gu, acth, 8*Cap*HID);
  gemm_wmma_f16<<<dim3(Dd/128, Cap/32, 8), 256, 0, stream>>>(
      acth, w2h, eout, nullptr, buf, Cap, HID, Dd, HID, Dd, Dd, Dd,
      (long long)Cap*HID, (long long)HID*Dd, (long long)Cap*Dd, (long long)Cap*Dd, 1.f);

  // --- collaboration rounds ---
  build_seq_k<<<(N*Dd + 255)/256, 256, 0, stream>>>(medi, eout, tidx, slotA, keepA, seq, Cap, N);
  for (int r = 0; r < 2; ++r) {
    rmsnorm_k<<<N3, 256, 0, stream>>>(seq, cn1, snormh, Dd);
    gemm_wmma_f16<<<dim3(Dd/128, N3/32, 1), 256, 0, stream>>>(
        snormh, mwqh, sqb, mbq, nullptr, N3, Dd, Dd, Dd, Dd, Dd, 0, 0,0,0,0, 1.f);
    gemm_wmma_f16<<<dim3(Dd/128, N3/32, 1), 256, 0, stream>>>(
        snormh, mwkh, skb, mbk, nullptr, N3, Dd, Dd, Dd, Dd, Dd, 0, 0,0,0,0, 1.f);
    gemm_wmma_f16<<<dim3(Dd/128, N3/32, 1), 256, 0, stream>>>(
        snormh, mwvh, svb, mbv, nullptr, N3, Dd, Dd, Dd, Dd, Dd, 0, 0,0,0,0, 1.f);
    collab_attn_k<<<(N*12 + 255)/256, 256, 0, stream>>>(sqb, skb, svb, sobh, N);
    gemm_wmma_f16<<<dim3(Dd/128, N3/32, 1), 256, 0, stream>>>(
        sobh, mwoh, seq, mbo, seq, N3, Dd, Dd, Dd, Dd, Dd, Dd, 0,0,0,0, 1.f);
    rmsnorm_k<<<N3, 256, 0, stream>>>(seq, cn2, snormh, Dd);
    gemm_wmma_f16<<<dim3(Dd/128, N3/32, 1), 256, 0, stream>>>(
        snormh, cf1h, sqb, nullptr, nullptr, N3, Dd, Dd, Dd, Dd, Dd, 0, 0,0,0,0, 1.f);
    gelu_k<<<((size_t)N3*Dd + 255)/256, 256, 0, stream>>>(sqb, sqh, N3*Dd);
    gemm_wmma_f16<<<dim3(Dd/128, N3/32, 1), 256, 0, stream>>>(
        sqh, cf2h, seq, nullptr, seq, N3, Dd, Dd, Dd, Dd, Dd, Dd, 0,0,0,0, 1.f);
  }

  // --- gated fuse + final projection into d_out ---
  gsel_k<<<N, 256, 0, stream>>>(seq, fgw, fgb, gsel);
  fuse_k<<<(N*Dd + 255)/256, 256, 0, stream>>>(seq, tprobs, gsel, fusedh, N);
  gemm_wmma_f16<<<dim3(Dd/128, N/32, 1), 256, 0, stream>>>(
      fusedh, moeoh, out, nullptr, x1, N, Dd, Dd, Dd, Dd, Dd, Dd, 0,0,0,0, 1.f);
}
